// GNNSimulator_26637387170146
// MI455X (gfx1250) — matrix operand order repaired and checked by reading, not compiled, not run on hardware
//
#include <hip/hip_runtime.h>

// ---------------------------------------------------------------------------
// GNN simulator forward pass for gfx1250 (MI455X), wave32 + WMMA f16.
// All GEMMs run on v_wmma_f32_16x16x32_f16. One wave computes a 16-row x
// 128-col output tile. Weights are pre-swizzled per-launch into the exact
// B-fragment lane layout; activations are staged in LDS for A-fragment loads.
// ---------------------------------------------------------------------------

typedef __attribute__((ext_vector_type(16))) _Float16 v16h;
typedef __attribute__((ext_vector_type(8)))  _Float16 v8h;
typedef __attribute__((ext_vector_type(8)))  float    v8f;

#define LN_EPS   1e-5f
#define NORM_EPS 1e-8f

struct MlpW {
    const _Float16 *w1, *w2, *w3;     // swizzled f16 weights
    const float    *b1, *b2, *b3;     // raw f32 biases
    const float    *g,  *be;          // LN affine (null if no LN)
};

// ---------------- WMMA fragment loads ----------------

// A fragment (16x32 f16) from LDS row-major activations.
// Per ISA 05_wmma: lanes 0-15 = row m, K in {k0..k0+7, k0+16..k0+23};
// lanes 16-31 = row m, K in {k0+8..k0+15, k0+24..k0+31}.
__device__ __forceinline__ v16h load_a(const _Float16* A, int stride, int k0, int lane) {
    int m  = lane & 15;
    int kh = (lane >> 4) * 8;
    const _Float16* p = A + m * stride + k0 + kh;
    v8h lo = *(const v8h*)(p);
    v8h hi = *(const v8h*)(p + 16);
    v16h r;
#pragma unroll
    for (int i = 0; i < 8; ++i) { r[i] = lo[i]; r[i + 8] = hi[i]; }
    return r;
}

// B fragment (32x16 f16): pre-swizzled, one 32B contiguous load per lane.
__device__ __forceinline__ v16h load_b(const _Float16* w, int frag, int lane) {
    return *(const v16h*)(w + (size_t)frag * 512 + lane * 16);
}

// Store accumulator tile to LDS (f16) with bias + ReLU.  C/D layout:
// element r of lane L -> row r + 8*(L>>4), col = tile*16 + (L&15).
__device__ __forceinline__ void store_relu(_Float16* O, const v8f* acc,
                                           const float* bias, int lane) {
    int rowoff = (lane >> 4) * 8, nl = lane & 15;
#pragma unroll
    for (int nt = 0; nt < 8; ++nt) {
        float b = bias[nt * 16 + nl];
#pragma unroll
        for (int r = 0; r < 8; ++r) {
            float v = acc[nt][r] + b;
            O[(rowoff + r) * 128 + nt * 16 + nl] = (_Float16)(v > 0.f ? v : 0.f);
        }
    }
}

// Two hidden ReLU GEMM layers: sA(K=NKB1*32) -> sB1(128) -> sB2(128).
// sB2 may alias sA (sA is dead after GEMM-1 reads it).
template <int NKB1>
__device__ __forceinline__ void mlp_hidden(const _Float16* sA, int aStride,
                                           _Float16* sB1, _Float16* sB2,
                                           const MlpW& W, int lane) {
    v8f z = {};
    v8f acc[8];
#pragma unroll
    for (int nt = 0; nt < 8; ++nt) acc[nt] = z;
    for (int kb = 0; kb < NKB1; ++kb) {
        v16h aF = load_a(sA, aStride, kb * 32, lane);
#pragma unroll
        for (int nt = 0; nt < 8; ++nt)
            acc[nt] = __builtin_amdgcn_wmma_f32_16x16x32_f16(
                false, aF, false, load_b(W.w1, kb * 8 + nt, lane),
                (short)0, acc[nt], false, false);
    }
    store_relu(sB1, acc, W.b1, lane);
    __builtin_amdgcn_wave_barrier();
#pragma unroll
    for (int nt = 0; nt < 8; ++nt) acc[nt] = z;
#pragma unroll
    for (int kb = 0; kb < 4; ++kb) {
        v16h aF = load_a(sB1, 128, kb * 32, lane);
#pragma unroll
        for (int nt = 0; nt < 8; ++nt)
            acc[nt] = __builtin_amdgcn_wmma_f32_16x16x32_f16(
                false, aF, false, load_b(W.w2, kb * 8 + nt, lane),
                (short)0, acc[nt], false, false);
    }
    store_relu(sB2, acc, W.b2, lane);
    __builtin_amdgcn_wave_barrier();
}

// Final GEMM (K=128) into accumulators, no activation.
template <int NT3>
__device__ __forceinline__ void gemm_last(const _Float16* sB2, const _Float16* w3,
                                          v8f* acc, int lane) {
    v8f z = {};
#pragma unroll
    for (int nt = 0; nt < NT3; ++nt) acc[nt] = z;
#pragma unroll
    for (int kb = 0; kb < 4; ++kb) {
        v16h aF = load_a(sB2, 128, kb * 32, lane);
#pragma unroll
        for (int nt = 0; nt < NT3; ++nt)
            acc[nt] = __builtin_amdgcn_wmma_f32_16x16x32_f16(
                false, aF, false, load_b(w3, kb * NT3 + nt, lane),
                (short)0, acc[nt], false, false);
    }
}

// LayerNorm row stats on the accumulator layout: adds bias b3, then per-row
// mean/rstd via half-wave shfl_xor reductions (row m lives in 16 lanes).
__device__ __forceinline__ void ln_stats8(v8f* acc, const float* b3, int nl,
                                          float* mu, float* rs) {
#pragma unroll
    for (int nt = 0; nt < 8; ++nt) {
        float b = b3[nt * 16 + nl];
#pragma unroll
        for (int r = 0; r < 8; ++r) acc[nt][r] += b;
    }
#pragma unroll
    for (int r = 0; r < 8; ++r) {
        float s = 0.f, ss = 0.f;
#pragma unroll
        for (int nt = 0; nt < 8; ++nt) { float v = acc[nt][r]; s += v; ss += v * v; }
#pragma unroll
        for (int m = 1; m < 16; m <<= 1) {
            s  += __shfl_xor(s,  m, 32);
            ss += __shfl_xor(ss, m, 32);
        }
        float m_ = s * (1.f / 128.f);
        float var = fmaxf(ss * (1.f / 128.f) - m_ * m_, 0.f);
        mu[r] = m_;
        rs[r] = rsqrtf(var + LN_EPS);
    }
}

// dataset-stats normalization helper: returns 1/std, outputs mean.
__device__ __forceinline__ float stat_rinv(const float* cnt, const float* sum,
                                           const float* ssq, int d, float& mean) {
    float c = fmaxf(cnt[0], 1.f);
    mean = sum[d] / c;
    float var = fmaxf(ssq[d] / c - mean * mean, 0.f);
    return 1.f / sqrtf(var + NORM_EPS);
}

// ---------------- kernels ----------------

// Swizzle one raw f32 weight matrix [Kraw x Ncols] into B-fragment layout.
// grid.x = NKB*NT frags, 32 threads. Pads K and N with zeros.
__global__ void k_prep(const float* __restrict__ W, int Kraw, int Ncols, int NT,
                       _Float16* __restrict__ out) {
    int frag = blockIdx.x, lane = threadIdx.x;
    int kb = frag / NT, nt = frag % NT;
    int n  = nt * 16 + (lane & 15);
    int k0 = kb * 32 + ((lane >> 4) * 8);
    v16h v;
#pragma unroll
    for (int i = 0; i < 16; ++i) {
        int k = k0 + (i & 7) + ((i >> 3) << 4);
        float x = (k < Kraw && n < Ncols) ? W[(size_t)k * Ncols + n] : 0.f;
        v[i] = (_Float16)x;
    }
    *(v16h*)(out + (size_t)frag * 512 + lane * 16) = v;
}

__global__ void k_zero(float* __restrict__ p, int n) {
    int i = blockIdx.x * 256 + threadIdx.x;
    if (i < n) p[i] = 0.f;
}

// Node encoder: [norm(x) | embed[type]] (32) -> MLP -> LN -> h32/h16.
__global__ void __launch_bounds__(128) k_enc_node(
    const float* __restrict__ x, const int* __restrict__ ptype,
    const float* __restrict__ embed,
    const float* cnt, const float* ssum, const float* ssq,
    MlpW W, float* __restrict__ h32, _Float16* __restrict__ h16, int N) {
    __shared__ alignas(16) _Float16 sA[4][16 * 32];
    __shared__ alignas(16) _Float16 sB1[4][16 * 128];
    __shared__ alignas(16) _Float16 sB2[4][16 * 128];
    int wave = threadIdx.x >> 5, lane = threadIdx.x & 31;
    int row0 = (blockIdx.x * 4 + wave) * 16;
    float mean = 0.f, rinv = 0.f;
    if (lane < 16) rinv = stat_rinv(cnt, ssum, ssq, lane, mean);
    for (int r = 0; r < 16; ++r) {
        int node = row0 + r;
        float v = 0.f;
        if (node < N) {
            if (lane < 16) v = (x[(size_t)node * 16 + lane] - mean) * rinv;
            else           v = embed[(size_t)ptype[node] * 16 + (lane - 16)];
        }
        sA[wave][r * 32 + lane] = (_Float16)v;
    }
    __builtin_amdgcn_wave_barrier();
    v8f acc[8];
    mlp_hidden<1>(sA[wave], 32, sB1[wave], sB2[wave], W, lane);
    gemm_last<8>(sB2[wave], W.w3, acc, lane);
    float mu[8], rs[8];
    int nl = lane & 15, rowoff = (lane >> 4) * 8;
    ln_stats8(acc, W.b3, nl, mu, rs);
#pragma unroll
    for (int nt = 0; nt < 8; ++nt) {
        int col = nt * 16 + nl;
        float gv = W.g[col], bev = W.be[col];
#pragma unroll
        for (int r = 0; r < 8; ++r) {
            int node = row0 + rowoff + r;
            float y = (acc[nt][r] - mu[r]) * rs[r] * gv + bev;
            if (node < N) {
                size_t o = (size_t)node * 128 + col;
                h32[o] = y; h16[o] = (_Float16)y;
            }
        }
    }
}

// Edge encoder: norm(edge_attr) (4, zero-padded to 32) -> MLP -> LN -> e32/e16.
__global__ void __launch_bounds__(128) k_enc_edge(
    const float* __restrict__ ea,
    const float* cnt, const float* ssum, const float* ssq,
    MlpW W, float* __restrict__ e32, _Float16* __restrict__ e16, int E) {
    __shared__ alignas(16) _Float16 sA[4][16 * 32];
    __shared__ alignas(16) _Float16 sB1[4][16 * 128];
    __shared__ alignas(16) _Float16 sB2[4][16 * 128];
    int wave = threadIdx.x >> 5, lane = threadIdx.x & 31;
    int row0 = (blockIdx.x * 4 + wave) * 16;
    float mean = 0.f, rinv = 0.f;
    if (lane < 4) rinv = stat_rinv(cnt, ssum, ssq, lane, mean);
    for (int r = 0; r < 16; ++r) {
        int er = row0 + r;
        float v = 0.f;
        if (er < E && lane < 4) v = (ea[(size_t)er * 4 + lane] - mean) * rinv;
        sA[wave][r * 32 + lane] = (_Float16)v;
    }
    __builtin_amdgcn_wave_barrier();
    v8f acc[8];
    mlp_hidden<1>(sA[wave], 32, sB1[wave], sB2[wave], W, lane);
    gemm_last<8>(sB2[wave], W.w3, acc, lane);
    float mu[8], rs[8];
    int nl = lane & 15, rowoff = (lane >> 4) * 8;
    ln_stats8(acc, W.b3, nl, mu, rs);
#pragma unroll
    for (int nt = 0; nt < 8; ++nt) {
        int col = nt * 16 + nl;
        float gv = W.g[col], bev = W.be[col];
#pragma unroll
        for (int r = 0; r < 8; ++r) {
            int er = row0 + rowoff + r;
            float y = (acc[nt][r] - mu[r]) * rs[r] * gv + bev;
            if (er < E) {
                size_t o = (size_t)er * 128 + col;
                e32[o] = y; e16[o] = (_Float16)y;
            }
        }
    }
}

// Edge message layer: gather [h16[s] | h16[r] | e16] (384) -> MLP -> LN,
// residual into e32/e16, atomic segment-sum of new e into agg[receiver].
__global__ void __launch_bounds__(96) k_edge(
    const int* __restrict__ eidx, const _Float16* __restrict__ h16,
    float* __restrict__ e32, _Float16* __restrict__ e16,
    float* __restrict__ agg, MlpW W, int E) {
    __shared__ alignas(16) _Float16 sA[3][16 * 384];   // GEMM-2 output aliases this
    __shared__ alignas(16) _Float16 sB1[3][16 * 128];
    int wave = threadIdx.x >> 5, lane = threadIdx.x & 31;
    int row0 = (blockIdx.x * 3 + wave) * 16;
    int sidx = 0, ridx = 0;
    if (lane < 16) {
        int er = row0 + lane;
        if (er < E) { sidx = eidx[er]; ridx = eidx[(size_t)E + er]; }
    }
    for (int r = 0; r < 16; ++r) {
        int er = row0 + r;
        int s  = __shfl(sidx, r, 32);
        int rc = __shfl(ridx, r, 32);
        _Float16* dst = &sA[wave][r * 384];
        if (er < E) {
            *(uint2*)(dst +       lane * 4) = *(const uint2*)(h16 + (size_t)s  * 128 + lane * 4);
            *(uint2*)(dst + 128 + lane * 4) = *(const uint2*)(h16 + (size_t)rc * 128 + lane * 4);
            *(uint2*)(dst + 256 + lane * 4) = *(const uint2*)(e16 + (size_t)er * 128 + lane * 4);
        } else {
            uint2 z; z.x = 0u; z.y = 0u;
            *(uint2*)(dst + lane * 4) = z;
            *(uint2*)(dst + 128 + lane * 4) = z;
            *(uint2*)(dst + 256 + lane * 4) = z;
        }
    }
    __builtin_amdgcn_wave_barrier();
    v8f acc[8];
    mlp_hidden<12>(sA[wave], 384, sB1[wave], sA[wave], W, lane);
    gemm_last<8>(sA[wave], W.w3, acc, lane);
    float mu[8], rs[8];
    int nl = lane & 15, rowoff = (lane >> 4) * 8;
    ln_stats8(acc, W.b3, nl, mu, rs);
#pragma unroll
    for (int nt = 0; nt < 8; ++nt) {
        int col = nt * 16 + nl;
        float gv = W.g[col], bev = W.be[col];
#pragma unroll
        for (int r = 0; r < 8; ++r) {
            int row = rowoff + r, er = row0 + row;
            int rc = __shfl(ridx, row, 32);
            float y = (acc[nt][r] - mu[r]) * rs[r] * gv + bev;
            if (er < E) {
                size_t o = (size_t)er * 128 + col;
                float enew = e32[o] + y;
                e32[o] = enew; e16[o] = (_Float16)enew;
                atomicAdd(&agg[(size_t)rc * 128 + col], enew);
            }
        }
    }
}

// Node update layer: [h16 | f16(agg)] (256) -> MLP -> LN, residual into h32/h16.
__global__ void __launch_bounds__(128) k_node(
    const float* __restrict__ agg, MlpW W,
    float* __restrict__ h32, _Float16* __restrict__ h16, int N) {
    __shared__ alignas(16) _Float16 sA[4][16 * 256];   // GEMM-2 output aliases this
    __shared__ alignas(16) _Float16 sB1[4][16 * 128];
    int wave = threadIdx.x >> 5, lane = threadIdx.x & 31;
    int row0 = (blockIdx.x * 4 + wave) * 16;
    for (int r = 0; r < 16; ++r) {
        int node = row0 + r;
        _Float16* dst = &sA[wave][r * 256];
        if (node < N) {
            *(uint2*)(dst + lane * 4) = *(const uint2*)(h16 + (size_t)node * 128 + lane * 4);
            float4 a4 = *(const float4*)(agg + (size_t)node * 128 + lane * 4);
            _Float16* d2 = dst + 128 + lane * 4;
            d2[0] = (_Float16)a4.x; d2[1] = (_Float16)a4.y;
            d2[2] = (_Float16)a4.z; d2[3] = (_Float16)a4.w;
        } else {
            uint2 z; z.x = 0u; z.y = 0u;
            *(uint2*)(dst + lane * 4) = z;
            *(uint2*)(dst + 128 + lane * 4) = z;
        }
    }
    __builtin_amdgcn_wave_barrier();
    v8f acc[8];
    mlp_hidden<8>(sA[wave], 256, sB1[wave], sA[wave], W, lane);
    gemm_last<8>(sA[wave], W.w3, acc, lane);
    float mu[8], rs[8];
    int nl = lane & 15, rowoff = (lane >> 4) * 8;
    ln_stats8(acc, W.b3, nl, mu, rs);
#pragma unroll
    for (int nt = 0; nt < 8; ++nt) {
        int col = nt * 16 + nl;
        float gv = W.g[col], bev = W.be[col];
#pragma unroll
        for (int r = 0; r < 8; ++r) {
            int node = row0 + rowoff + r;
            float y = (acc[nt][r] - mu[r]) * rs[r] * gv + bev;
            if (node < N) {
                size_t o = (size_t)node * 128 + col;
                float hnew = h32[o] + y;
                h32[o] = hnew; h16[o] = (_Float16)hnew;
            }
        }
    }
}

// Decoder: h (128) -> MLP -> 3 outputs, inverse-normalized with output stats.
__global__ void __launch_bounds__(128) k_dec(
    const _Float16* __restrict__ h16, MlpW W,
    const float* cnt, const float* ssum, const float* ssq,
    float* __restrict__ out, int N) {
    __shared__ alignas(16) _Float16 sA[4][16 * 128];   // GEMM-2 output aliases this
    __shared__ alignas(16) _Float16 sB1[4][16 * 128];
    int wave = threadIdx.x >> 5, lane = threadIdx.x & 31;
    int row0 = (blockIdx.x * 4 + wave) * 16;
    for (int r = 0; r < 16; ++r) {
        int node = row0 + r;
        _Float16* dst = &sA[wave][r * 128];
        if (node < N) {
            *(uint2*)(dst + lane * 4) = *(const uint2*)(h16 + (size_t)node * 128 + lane * 4);
        } else {
            uint2 z; z.x = 0u; z.y = 0u;
            *(uint2*)(dst + lane * 4) = z;
        }
    }
    __builtin_amdgcn_wave_barrier();
    v8f acc[1];
    mlp_hidden<4>(sA[wave], 128, sB1[wave], sA[wave], W, lane);
    gemm_last<1>(sA[wave], W.w3, acc, lane);
    int nl = lane & 15, rowoff = (lane >> 4) * 8;
    float b3v = 0.f, meanc = 0.f, stdc = 0.f;
    if (nl < 3) {
        b3v = W.b3[nl];
        float c = fmaxf(cnt[0], 1.f);
        meanc = ssum[nl] / c;
        float var = fmaxf(ssq[nl] / c - meanc * meanc, 0.f);
        stdc = sqrtf(var + NORM_EPS);
    }
#pragma unroll
    for (int r = 0; r < 8; ++r) {
        int node = row0 + rowoff + r;
        if (node < N && nl < 3)
            out[(size_t)node * 3 + nl] = (acc[0][r] + b3v) * stdc + meanc;
    }
}

// ---------------- host driver ----------------

extern "C" void kernel_launch(void* const* d_in, const int* in_sizes, int n_in,
                              void* d_out, int out_size, void* d_ws, size_t ws_size,
                              hipStream_t stream) {
    (void)n_in; (void)out_size;
    const float* x      = (const float*)d_in[0];
    const float* eattr  = (const float*)d_in[1];
    const int*   eidx   = (const int*)d_in[2];
    const int*   ptype  = (const int*)d_in[3];
    const float* embed  = (const float*)d_in[4];
    const int N = in_sizes[0] / 16;
    const int E = in_sizes[1] / 4;

    // ---- carve workspace ----
    size_t off = 0;
    auto carve = [&](size_t bytes) -> void* {
        off = (off + 255) & ~(size_t)255;
        void* p = (char*)d_ws + off;
        off += bytes;
        return p;
    };
    float*     h32  = (float*)carve((size_t)N * 128 * 4);
    _Float16*  h16  = (_Float16*)carve((size_t)N * 128 * 2);
    float*     e32  = (float*)carve((size_t)E * 128 * 4);
    _Float16*  e16  = (_Float16*)carve((size_t)E * 128 * 2);
    float*     agg  = (float*)carve((size_t)N * 128 * 4);
    _Float16*  wbuf = (_Float16*)carve((size_t)4 * 1024 * 1024);
    if (off > ws_size) return;   // workspace too small: bail deterministically

    // ---- pre-swizzle all weights into WMMA B-fragment layout ----
    _Float16* wcur = wbuf;
    auto prep = [&](const float* W, int Kraw, int Ncols, int NT) -> const _Float16* {
        int NKB = (Kraw + 31) / 32;
        _Float16* dst = wcur;
        k_prep<<<dim3(NKB * NT), dim3(32), 0, stream>>>(W, Kraw, Ncols, NT, dst);
        wcur += (size_t)NKB * NT * 512;
        return dst;
    };
    auto mk = [&](int i, int NT3, int Ncols3, bool ln) -> MlpW {
        MlpW w;
        int K1 = in_sizes[i] / 128;
        w.w1 = prep((const float*)d_in[i],     K1,  128,    8);
        w.b1 = (const float*)d_in[i + 1];
        w.w2 = prep((const float*)d_in[i + 2], 128, 128,    8);
        w.b2 = (const float*)d_in[i + 3];
        w.w3 = prep((const float*)d_in[i + 4], 128, Ncols3, NT3);
        w.b3 = (const float*)d_in[i + 5];
        w.g  = ln ? (const float*)d_in[i + 6] : nullptr;
        w.be = ln ? (const float*)d_in[i + 7] : nullptr;
        return w;
    };
    MlpW wEncN = mk(5,  8, 128, true);
    MlpW wEncE = mk(13, 8, 128, true);
    MlpW wDec  = mk(21, 1, 3,   false);
    MlpW wEdge[10], wNode[10];
    for (int i = 0; i < 10; ++i) {
        wEdge[i] = mk(27 + 16 * i,     8, 128, true);
        wNode[i] = mk(27 + 16 * i + 8, 8, 128, true);
    }
    const float* stN[3] = { (const float*)d_in[187], (const float*)d_in[188], (const float*)d_in[189] };
    const float* stE[3] = { (const float*)d_in[190], (const float*)d_in[191], (const float*)d_in[192] };
    const float* stO[3] = { (const float*)d_in[193], (const float*)d_in[194], (const float*)d_in[195] };

    // ---- encoders ----
    int gN = (N + 63) / 64;
    int gE4 = (E + 63) / 64;
    int gE3 = (E + 47) / 48;
    k_enc_node<<<dim3(gN), dim3(128), 0, stream>>>(x, ptype, embed,
        stN[0], stN[1], stN[2], wEncN, h32, h16, N);
    k_enc_edge<<<dim3(gE4), dim3(128), 0, stream>>>(eattr,
        stE[0], stE[1], stE[2], wEncE, e32, e16, E);

    // ---- message-passing layers ----
    int nAgg = N * 128;
    for (int l = 0; l < 10; ++l) {
        k_zero<<<dim3((nAgg + 255) / 256), dim3(256), 0, stream>>>(agg, nAgg);
        k_edge<<<dim3(gE3), dim3(96), 0, stream>>>(eidx, h16, e32, e16, agg, wEdge[l], E);
        k_node<<<dim3(gN), dim3(128), 0, stream>>>(agg, wNode[l], h32, h16, N);
    }

    // ---- decoder ----
    k_dec<<<dim3(gN), dim3(128), 0, stream>>>(h16, wDec,
        stO[0], stO[1], stO[2], (float*)d_out, N);
}
